// M3Gnet_18880676233721
// MI455X (gfx1250) — matrix operand order repaired and checked by reading, not compiled, not run on hardware
//
#include <hip/hip_runtime.h>
#include <hip/hip_bf16.h>
#include <math.h>

#define UNITS   128
#define MAXN    4
#define MAXL    4
#define LNCH    16        // MAX_L * MAX_N
#define CUT     5.0f
#define TBCUT   4.0f

typedef __attribute__((ext_vector_type(16))) _Float16 v16h;
typedef __attribute__((ext_vector_type(8)))  float    v8f;

__device__ __constant__ float c_sbz[16] = {
    3.14159265358979f, 6.28318530717959f, 9.42477796076938f, 12.5663706143592f,
    4.49340945790906f, 7.72525183693771f, 10.9041216594289f, 14.0661939128315f,
    5.76345919689455f, 9.09501133047635f, 12.3229409705666f, 15.5146030108867f,
    6.98793200050052f, 10.4171185473794f, 13.6980231532502f, 16.9236212852143f};

// fast activations: v_exp_f32 + v_rcp_f32 (no IEEE div sequences)
__device__ __forceinline__ float sigmf(float x){
    return __builtin_amdgcn_rcpf(1.0f + __expf(-x));
}
__device__ __forceinline__ float siluf(float x){
    return x * __builtin_amdgcn_rcpf(1.0f + __expf(-x));
}

// ---------------------------------------------------------------- utility
__global__ void zero_kernel(float* p, long long n){
    long long i = (long long)blockIdx.x * blockDim.x + threadIdx.x;
    if (i < n) p[i] = 0.0f;
}

__global__ void scan_kernel(const int* nb, const int* nt, int G, int* bstart, int* tstart){
    if (blockIdx.x == 0 && threadIdx.x == 0){
        int b = 0, t = 0;
        for (int g = 0; g < G; ++g){ bstart[g] = b; tstart[g] = t; b += nb[g]; t += nt[g]; }
    }
}

__global__ void init_h_kernel(const int* attr, const float* emb, float* h, _Float16* h16, int N){
    int i = blockIdx.x * blockDim.x + threadIdx.x;
    if (i >= N * UNITS) return;
    int n = i >> 7, u = i & 127;
    float v = emb[attr[n] * UNITS + u];
    h[i] = v; h16[i] = (_Float16)v;
}

// ------------------------------------------------- weight repack (f32 -> f16 WMMA-B order)
// out[((ntile*KT + kt)*32 + lane)*16 + j] = W[k*128 + n]
//   n = ntile*16 + (lane&15),  k = kt*32 + (j<8 ? (lane>>4)*8 + j : 16 + (lane>>4)*8 + (j-8))    (ISA 16-bit B layout)
__global__ void repack_w_kernel(const float* __restrict__ W, _Float16* __restrict__ out, int Kdim){
    int KT = Kdim >> 5;
    int total = 8 * KT * 32 * 16;
    int i = blockIdx.x * blockDim.x + threadIdx.x;
    if (i >= total) return;
    int j    = i & 15;
    int lane = (i >> 4) & 31;
    int kt   = (i >> 9) % KT;
    int nt   = (i >> 9) / KT;
    int n = nt * 16 + (lane & 15);
    int k = kt * 32 + (j < 8 ? (lane >> 4) * 8 + j : 16 + (lane >> 4) * 8 + (j - 8));
    out[i] = (_Float16)W[k * UNITS + n];
}

// ------------------------------------------------- geometry + radial basis
__global__ void geom_kernel(const float* __restrict__ pos, const float* __restrict__ cell,
                            const float* __restrict__ pbc, const int* __restrict__ src,
                            const int* __restrict__ dst, const int* __restrict__ batch,
                            float* __restrict__ evec, float* __restrict__ elen,
                            float* __restrict__ rbf, float* __restrict__ fc, int E){
    int e = blockIdx.x * blockDim.x + threadIdx.x;
    if (e >= E) return;
    int s = src[e], d = dst[e];
    int g = batch[s];
    float off0 = pbc[e*3+0], off1 = pbc[e*3+1], off2 = pbc[e*3+2];
    float v[3];
    #pragma unroll
    for (int j = 0; j < 3; ++j){
        float mv = off0 * cell[g*9 + 0*3 + j] + off1 * cell[g*9 + 1*3 + j] + off2 * cell[g*9 + 2*3 + j];
        v[j] = pos[s*3+j] - (pos[d*3+j] + mv);
    }
    float r2 = v[0]*v[0] + v[1]*v[1] + v[2]*v[2];
    float r  = sqrtf(r2);
    evec[e*3+0]=v[0]; evec[e*3+1]=v[1]; evec[e*3+2]=v[2];
    elen[e] = r;
    // smooth bessel
    float rr = fmaxf(r, 1e-9f);
    float fs[MAXN];
    const float pi = 3.14159265358979f;
    #pragma unroll
    for (int n = 0; n < MAXN; ++n){
        float a = (float)(n+1), b = (float)(n+2);
        float pref = ((n & 1) ? -1.0f : 1.0f) * 1.41421356237f * pi / (CUT * sqrtf(CUT))
                     * (a * b / sqrtf(a*a + b*b));
        float x1 = a * rr / CUT, x2 = b * rr / CUT;
        float s1 = __sinf(pi*x1) / (pi*x1);
        float s2 = __sinf(pi*x2) / (pi*x2);
        fs[n] = pref * (s1 + s2);
    }
    float gprev = fs[0], dprev = 1.0f;
    rbf[e*4+0] = gprev;
    #pragma unroll
    for (int n = 1; n < MAXN; ++n){
        float fn = (float)n, fn1 = fn+1.0f, fn2 = fn+2.0f;
        float en = fn*fn*fn2*fn2 / (4.0f*fn1*fn1*fn1*fn1 + 1.0f);
        float dn = 1.0f - en / dprev;
        float gn = (fs[n] + sqrtf(en/dprev) * gprev) / sqrtf(dn);
        rbf[e*4+n] = gn;
        gprev = gn; dprev = dn;
    }
    // poly cutoff (TB)
    float t = r / TBCUT;
    float t2 = t*t, t3 = t2*t;
    fc[e] = (r < TBCUT) ? (1.0f - 6.0f*t3*t2 + 15.0f*t2*t2 - 10.0f*t3) : 0.0f;
}

__global__ void e_init_kernel(const float* __restrict__ rbf, const float* __restrict__ enc,
                              float* __restrict__ eout, _Float16* __restrict__ e16, long long EU){
    long long i = (long long)blockIdx.x * blockDim.x + threadIdx.x;
    if (i >= EU) return;
    int e = (int)(i >> 7), u = (int)(i & 127);
    float acc = 0.0f;
    #pragma unroll
    for (int q = 0; q < 4; ++q) acc += rbf[e*4+q] * enc[q*UNITS+u];
    float v = siluf(acc);
    eout[i] = v; e16[i] = (_Float16)v;
}

// ------------------------------------------------- three-body features
__global__ void tb_kernel(const int* __restrict__ tbi_in, const int* __restrict__ bstart,
                          const int* __restrict__ tstart, int G,
                          const float* __restrict__ evec, const float* __restrict__ elen,
                          const int* __restrict__ dst, const float* __restrict__ fc,
                          float* __restrict__ tb, int* __restrict__ tbi0,
                          int* __restrict__ kidx, float* __restrict__ wts, int T){
    int t = blockIdx.x * blockDim.x + threadIdx.x;
    if (t >= T) return;
    int g = 0;
    while (g + 1 < G && t >= tstart[g+1]) ++g;
    int i0 = tbi_in[t*2+0] + bstart[g];
    int i1 = tbi_in[t*2+1] + bstart[g];
    tbi0[t] = i0;
    kidx[t] = dst[i1];
    wts[t]  = fc[i0] * fc[i1];
    float rij = elen[i0], rik = elen[i1];
    float dp = evec[i0*3]*evec[i1*3] + evec[i0*3+1]*evec[i1*3+1] + evec[i0*3+2]*evec[i1*3+2];
    float c = dp / (rij * rik);
    c = fminf(fmaxf(c, -1.0f + 1e-7f), 1.0f - 1e-7f);
    float r = fmaxf(rik, 1e-9f);
    float P[4];
    P[0] = 1.0f; P[1] = c; P[2] = 0.5f*(3.0f*c*c - 1.0f); P[3] = 0.5f*(5.0f*c*c*c - 3.0f*c);
    const float inv4pi = 0.0795774715459477f; // 1/(4*pi)
    #pragma unroll
    for (int l = 0; l < MAXL; ++l){
        float yl = sqrtf((2.0f*l + 1.0f) * inv4pi) * P[l];
        #pragma unroll
        for (int n = 0; n < MAXN; ++n){
            float x = c_sbz[l*4+n] * r / CUT;
            float sx = __sinf(x), cx = __cosf(x);
            float ix = __builtin_amdgcn_rcpf(x), ix2 = ix*ix;
            float jl;
            if      (l == 0) jl = sx * ix;
            else if (l == 1) jl = sx * ix2 - cx * ix;
            else if (l == 2) jl = (3.0f*ix2*ix - ix) * sx - 3.0f*ix2 * cx;
            else             jl = (15.0f*ix2*ix2 - 6.0f*ix2) * sx - (15.0f*ix2*ix - ix) * cx;
            tb[t*LNCH + l*MAXN + n] = jl * yl;
        }
    }
}

// ------------------------------------------------- per-block scalar kernels
__global__ void gatek_kernel(const float* __restrict__ h, const float* __restrict__ W,
                             const float* __restrict__ b, float* __restrict__ gk, int N){
    int i = blockIdx.x * blockDim.x + threadIdx.x;
    if (i >= N * LNCH) return;
    int n = i >> 4, j = i & 15;
    float acc = b[j];
    for (int k = 0; k < UNITS; ++k) acc += h[n*UNITS+k] * W[k*LNCH+j];
    gk[i] = sigmf(acc);
}

__global__ void msg_kernel(const float* __restrict__ tb, const int* __restrict__ tbi0,
                           const int* __restrict__ kidx, const float* __restrict__ wts,
                           const float* __restrict__ gk, float* __restrict__ nb, int T){
    int t = blockIdx.x * blockDim.x + threadIdx.x;
    if (t >= T) return;
    int i0 = tbi0[t], k = kidx[t];
    float w = wts[t];
    #pragma unroll
    for (int c = 0; c < LNCH; ++c)
        atomicAdd(&nb[i0*LNCH + c], tb[t*LNCH + c] * gk[k*LNCH + c] * w);
}

__global__ void e_addnb_kernel(float* __restrict__ e, _Float16* __restrict__ e16,
                               const float* __restrict__ nb,
                               const float* __restrict__ W, long long EU){
    long long i = (long long)blockIdx.x * blockDim.x + threadIdx.x;
    if (i >= EU) return;
    int row = (int)(i >> 7), u = (int)(i & 127);
    float acc = 0.0f;
    #pragma unroll
    for (int c = 0; c < LNCH; ++c) acc += nb[row*LNCH + c] * W[c*UNITS + u];
    float v = e[i] + acc;
    e[i] = v; e16[i] = (_Float16)v;
}

__global__ void add_kernel(float* __restrict__ a, _Float16* __restrict__ a16,
                           const float* __restrict__ b, long long n){
    long long i = (long long)blockIdx.x * blockDim.x + threadIdx.x;
    if (i < n){ float v = a[i] + b[i]; a[i] = v; a16[i] = (_Float16)v; }
}

// ------------------------------------------------- gated2 via WMMA (the hot path)
#define FST 392   // feat LDS stride (halves): 384 + pad, 784B row stride (16B aligned)
#define HST 136   // hidden LDS stride (halves)

__global__ void __launch_bounds__(256)
gated2_wmma_kernel(const _Float16* __restrict__ h16, const _Float16* __restrict__ e16,
                   const int* __restrict__ src, const int* __restrict__ dst,
                   const _Float16* __restrict__ W1p, const float* __restrict__ b1,
                   const _Float16* __restrict__ W2p, const float* __restrict__ b2,
                   const _Float16* __restrict__ G1p, const float* __restrict__ c1,
                   const _Float16* __restrict__ G2p, const float* __restrict__ c2,
                   const float* __restrict__ rbf, const float* __restrict__ projw,
                   float* __restrict__ e_out, _Float16* __restrict__ e16_out,
                   float* __restrict__ hacc, int mode, int E, int numTiles){
    __shared__ _Float16 sFeat[16*FST];
    __shared__ _Float16 sHidM[16*HST];
    __shared__ _Float16 sHidG[16*HST];
    __shared__ int sSrc[16], sDst[16];

    int t    = threadIdx.x;
    int lane = t & 31, wave = t >> 5;
    int half = lane >> 4, lm = lane & 15;
    int col  = wave * 16 + lm;              // output column owned by this lane

    const v16h* B1m = (const v16h*)W1p;
    const v16h* B1g = (const v16h*)G1p;
    const v16h* B2m = (const v16h*)W2p;
    const v16h* B2g = (const v16h*)G2p;
    float bias1m = b1[col], bias1g = c1[col];
    float bias2m = b2[col], bias2g = c2[col];

    // persistent tile loop: weight slices stay hot in WGP$ / registers
    for (int tile = blockIdx.x; tile < numTiles; tile += gridDim.x){
        int base = tile * 16;
        if (t < 16){
            int r = base + t; if (r >= E) r = E - 1;
            sSrc[t] = src[r]; sDst[t] = dst[r];
        }
        __syncthreads();

        // Build feat tile [16 x 384] f16 = concat(h16[src], h16[dst], e16) via 16B chunks
        for (int i = t; i < 768; i += 256){     // 16 rows * 3 segs * 16 chunks
            int row = i / 48;
            int rem = i - row * 48;
            int seg = rem >> 4, idx = rem & 15;
            int erow = base + row; if (erow >= E) erow = E - 1;
            const uint4* gp;
            if      (seg == 0) gp = (const uint4*)(h16 + (size_t)sSrc[row]*UNITS) + idx;
            else if (seg == 1) gp = (const uint4*)(h16 + (size_t)sDst[row]*UNITS) + idx;
            else               gp = (const uint4*)(e16 + (size_t)erow*UNITS) + idx;
            *reinterpret_cast<uint4*>(&sFeat[row*FST + seg*128 + idx*8]) = *gp;
        }
        __syncthreads();

        // prefetch next tile's gather rows (global_prefetch_b8) to hide latency under WMMA
        int ntile = tile + gridDim.x;
        if (ntile < numTiles && t < 16){
            int r = ntile * 16 + t; if (r >= E) r = E - 1;
            __builtin_prefetch(h16 + (size_t)src[r]*UNITS, 0, 3);
            __builtin_prefetch(h16 + (size_t)dst[r]*UNITS, 0, 3);
            __builtin_prefetch(e16 + (size_t)r*UNITS, 0, 3);
        }

        // ---- layer 1: 16x384 @ 384x128, main + gate
        v8f aM, aG;
        #pragma unroll
        for (int v = 0; v < 8; ++v){ aM[v] = bias1m; aG[v] = bias1g; }
        #pragma unroll 4
        for (int kt = 0; kt < 12; ++kt){
            v16h a;
            int kb = kt*32 + half*8;
            int rb = lm * FST;
            #pragma unroll
            for (int j = 0; j < 8; ++j) a[j]   = sFeat[rb + kb + j];       // A layout: K=half*8+j
            #pragma unroll
            for (int j = 0; j < 8; ++j) a[8+j] = sFeat[rb + 16 + kb + j];  //           K=16+half*8+j
            v16h bm = B1m[(wave*12 + kt)*32 + lane];
            aM = __builtin_amdgcn_wmma_f32_16x16x32_f16(false, a, false, bm, (short)0, aM, false, false);
            v16h bg = B1g[(wave*12 + kt)*32 + lane];
            aG = __builtin_amdgcn_wmma_f32_16x16x32_f16(false, a, false, bg, (short)0, aG, false, false);
        }
        // silu -> f16 hidden in LDS (C layout: VGPR v -> row v + 8*half)
        #pragma unroll
        for (int v = 0; v < 8; ++v){
            int row = v + half*8;
            sHidM[row*HST + col] = (_Float16)siluf(aM[v]);
            sHidG[row*HST + col] = (_Float16)siluf(aG[v]);
        }
        __syncthreads();

        // ---- layer 2: 16x128 @ 128x128, main + gate
        v8f dM, dG;
        #pragma unroll
        for (int v = 0; v < 8; ++v){ dM[v] = bias2m; dG[v] = bias2g; }
        #pragma unroll
        for (int kt = 0; kt < 4; ++kt){
            v16h am, ag;
            int kb = kt*32 + half*8;
            int rb = lm * HST;
            #pragma unroll
            for (int j = 0; j < 8; ++j){ am[j]   = sHidM[rb + kb + j];      ag[j]   = sHidG[rb + kb + j]; }
            #pragma unroll
            for (int j = 0; j < 8; ++j){ am[8+j] = sHidM[rb + 16 + kb + j]; ag[8+j] = sHidG[rb + 16 + kb + j]; }
            v16h bm = B2m[(wave*4 + kt)*32 + lane];
            dM = __builtin_amdgcn_wmma_f32_16x16x32_f16(false, am, false, bm, (short)0, dM, false, false);
            v16h bg = B2g[(wave*4 + kt)*32 + lane];
            dG = __builtin_amdgcn_wmma_f32_16x16x32_f16(false, ag, false, bg, (short)0, dG, false, false);
        }

        // ---- epilogue: silu(main) * sigmoid(gate) * (rbf @ projw), scatter
        #pragma unroll
        for (int v = 0; v < 8; ++v){
            int row  = v + half*8;
            int erow = base + row;
            if (erow >= E) continue;
            float p = 0.0f;
            #pragma unroll
            for (int q = 0; q < 4; ++q) p += rbf[erow*4+q] * projw[q*UNITS + col];
            float val = siluf(dM[v]) * sigmf(dG[v]) * p;
            if (mode == 0){
                size_t o = (size_t)erow*UNITS + col;
                float nv = e_out[o] + val;
                e_out[o] = nv; e16_out[o] = (_Float16)nv;
            } else {
                atomicAdd(&hacc[sSrc[row]*UNITS + col], val);
            }
        }
        __syncthreads();   // protect LDS before next iteration
    }
}

// ------------------------------------------------- final gated3 readout
__global__ void __launch_bounds__(128)
final_kernel(const float* __restrict__ h, const int* __restrict__ attr,
             const int* __restrict__ batch,
             const float* W1, const float* b1, const float* W2, const float* b2,
             const float* W3, const float* b3,
             const float* G1, const float* c1, const float* G2, const float* c2,
             const float* G3, const float* c3,
             const float* scale, const float* shift, float* out, int N){
    int a = blockIdx.x; if (a >= N) return;
    int j = threadIdx.x;
    __shared__ float xs[128], m1[128], g1s[128], m2[128], g2s[128], rm[128], rg[128];
    xs[j] = h[a*UNITS + j];
    __syncthreads();
    float am = b1[j], ag = c1[j];
    for (int k = 0; k < UNITS; ++k){ am += xs[k]*W1[k*UNITS+j]; ag += xs[k]*G1[k*UNITS+j]; }
    m1[j] = siluf(am); g1s[j] = siluf(ag);
    __syncthreads();
    am = b2[j]; ag = c2[j];
    for (int k = 0; k < UNITS; ++k){ am += m1[k]*W2[k*UNITS+j]; ag += g1s[k]*G2[k*UNITS+j]; }
    m2[j] = siluf(am); g2s[j] = siluf(ag);
    __syncthreads();
    rm[j] = m2[j]*W3[j]; rg[j] = g2s[j]*G3[j];
    __syncthreads();
    if (j == 0){
        float sm = b3[0], sg = c3[0];
        for (int k = 0; k < UNITS; ++k){ sm += rm[k]; sg += rg[k]; }
        int z = attr[a];
        float ei = sm * sigmf(sg);
        ei = ei * scale[z] + shift[z];
        atomicAdd(&out[batch[a]], ei);
    }
}

// =================================================================== launch
extern "C" void kernel_launch(void* const* d_in, const int* in_sizes, int n_in,
                              void* d_out, int out_size, void* d_ws, size_t ws_size,
                              hipStream_t stream){
    const float* atom_pos = (const float*)d_in[0];
    const float* cell     = (const float*)d_in[1];
    const float* pbc      = (const float*)d_in[2];
    const int*   attr     = (const int*)  d_in[3];
    const int*   eidx     = (const int*)  d_in[4];
    const int*   tbi_in   = (const int*)  d_in[5];
    const int*   num_tb   = (const int*)  d_in[6];
    const int*   num_bond = (const int*)  d_in[7];
    const int*   batch    = (const int*)  d_in[11];

    const int N = in_sizes[0] / 3;
    const int E = in_sizes[4] / 2;
    const int T = in_sizes[5] / 2;
    const int G = in_sizes[7];
    const int* src = eidx;
    const int* dst = eidx + E;

    auto P = [&](int i){ return (const float*)d_in[i]; };
    // params flattening: 12 atom_emb, 13 edge_enc, blocks at 14 (21 arrays each),
    // final_main 98..103, final_gate 104..109, scale 110, shift 111
    const int PB = 14;

    // ---------------- workspace carve-out
    char* ws = (char*)d_ws;
    size_t off = 0;
    auto carve = [&](size_t bytes)->char*{
        char* p = ws + off;
        off = (off + bytes + 255) & ~(size_t)255;
        return p;
    };
    float*    h     = (float*)   carve((size_t)N*UNITS*4);
    _Float16* h16   = (_Float16*)carve((size_t)N*UNITS*2);
    float*    hacc  = (float*)   carve((size_t)N*UNITS*4);
    float*    e     = (float*)   carve((size_t)E*UNITS*4);
    _Float16* e16   = (_Float16*)carve((size_t)E*UNITS*2);
    float*    evec  = (float*)   carve((size_t)E*3*4);
    float*    elen  = (float*)   carve((size_t)E*4);
    float*    rbf   = (float*)   carve((size_t)E*4*4);
    float*    fc    = (float*)   carve((size_t)E*4);
    float*    tb    = (float*)   carve((size_t)T*LNCH*4);
    int*      tbi0  = (int*)     carve((size_t)T*4);
    int*      kidx  = (int*)     carve((size_t)T*4);
    float*    wts   = (float*)   carve((size_t)T*4);
    float*    gk    = (float*)   carve((size_t)N*LNCH*4);
    float*    nb    = (float*)   carve((size_t)E*LNCH*4);
    int*      bstart= (int*)     carve((size_t)G*4);
    int*      tstart= (int*)     carve((size_t)G*4);
    _Float16* pk[4][8];
    static const int widx[8] = {5, 7, 9, 11, 13, 15, 17, 19};   // eW1,eW2,eG1,eG2,nW1,nW2,nG1,nG2
    static const int wk[8]   = {384,128,384,128,384,128,384,128};
    for (int b = 0; b < 4; ++b)
        for (int m = 0; m < 8; ++m)
            pk[b][m] = (_Float16*)carve((size_t)wk[m]*UNITS*2);

    // ---------------- one-time prep
    for (int b = 0; b < 4; ++b)
        for (int m = 0; m < 8; ++m){
            int total = 128 * wk[m];   // 8*(K/32)*32*16
            repack_w_kernel<<<(total+255)/256, 256, 0, stream>>>(P(PB + b*21 + widx[m]), pk[b][m], wk[m]);
        }
    scan_kernel<<<1, 1, 0, stream>>>(num_bond, num_tb, G, bstart, tstart);
    init_h_kernel<<<(N*UNITS+255)/256, 256, 0, stream>>>(attr, P(12), h, h16, N);
    geom_kernel<<<(E+255)/256, 256, 0, stream>>>(atom_pos, cell, pbc, src, dst, batch,
                                                 evec, elen, rbf, fc, E);
    long long EU = (long long)E * UNITS;
    e_init_kernel<<<(int)((EU+255)/256), 256, 0, stream>>>(rbf, P(13), e, e16, EU);
    tb_kernel<<<(T+255)/256, 256, 0, stream>>>(tbi_in, bstart, tstart, G, evec, elen,
                                               dst, fc, tb, tbi0, kidx, wts, T);

    // ---------------- 4 message-passing blocks
    int numTiles = (E + 15) / 16;
    int g2grid = numTiles < 2048 ? numTiles : 2048;   // persistent blocks
    for (int b = 0; b < 4; ++b){
        const int base = PB + b*21;
        gatek_kernel<<<(N*LNCH+255)/256, 256, 0, stream>>>(h, P(base+0), P(base+1), gk, N);
        zero_kernel<<<(int)(((long long)E*LNCH+255)/256), 256, 0, stream>>>(nb, (long long)E*LNCH);
        msg_kernel<<<(T+255)/256, 256, 0, stream>>>(tb, tbi0, kidx, wts, gk, nb, T);
        e_addnb_kernel<<<(int)((EU+255)/256), 256, 0, stream>>>(e, e16, nb, P(base+2), EU);
        // edge update: feat=[h_src,h_dst,e] -> gated2 -> * (rbf@e0), += e
        gated2_wmma_kernel<<<g2grid, 256, 0, stream>>>(h16, e16, src, dst,
            pk[b][0], P(base+6),  pk[b][1], P(base+8),
            pk[b][2], P(base+10), pk[b][3], P(base+12),
            rbf, P(base+3), e, e16, hacc, 0, E, numTiles);
        // node update: feat with updated e -> gated2 -> * (rbf@n0), segment-sum into h
        zero_kernel<<<(int)(((long long)N*UNITS+255)/256), 256, 0, stream>>>(hacc, (long long)N*UNITS);
        gated2_wmma_kernel<<<g2grid, 256, 0, stream>>>(h16, e16, src, dst,
            pk[b][4], P(base+14), pk[b][5], P(base+16),
            pk[b][6], P(base+18), pk[b][7], P(base+20),
            rbf, P(base+4), e, e16, hacc, 1, E, numTiles);
        add_kernel<<<(int)(((long long)N*UNITS+255)/256), 256, 0, stream>>>(h, h16, hacc, (long long)N*UNITS);
    }

    // ---------------- readout
    zero_kernel<<<1, 256, 0, stream>>>((float*)d_out, out_size);
    final_kernel<<<N, 128, 0, stream>>>(h, attr, batch,
        P(98), P(99), P(100), P(101), P(102), P(103),
        P(104), P(105), P(106), P(107), P(108), P(109),
        P(110), P(111), (float*)d_out, N);
}
